// LongformerSelfAttention_41669772705891
// MI455X (gfx1250) — matrix operand order repaired and checked by reading, not compiled, not run on hardware
//
#include <hip/hip_runtime.h>
#include <hip/hip_bf16.h>

// Problem constants (Longformer local attention)
#define BB 2
#define SS 4096
#define EE 768
#define HH 12
#define DD 64
#define WW 256
#define NCHUNK 16      // SS / WW
#define NQK 2304       // 3*EE packed output dim for fused QKV GEMM

typedef __bf16 bf16_t;
typedef __attribute__((ext_vector_type(8)))  __bf16 v8bf;
typedef __attribute__((ext_vector_type(16))) __bf16 v16bf;
typedef __attribute__((ext_vector_type(8)))  float  v8f;
typedef __attribute__((ext_vector_type(4)))  int    v4i;

// Address-space qualified int4 for the async global->LDS builtin
typedef __attribute__((address_space(1))) v4i gas_v4i;   // global (AS1)
typedef __attribute__((address_space(3))) v4i las_v4i;   // LDS    (AS3)

// Async global->LDS staging (CDNA5 ASYNCcnt path), with sync fallback.
#if defined(__gfx1250__) && __has_builtin(__builtin_amdgcn_global_load_async_to_lds_b128) && __has_builtin(__builtin_amdgcn_s_wait_asynccnt)
#define USE_ASYNC_LDS 1
#else
#define USE_ASYNC_LDS 0
#endif

__device__ __forceinline__ void stage16B(const unsigned short* gsrc, unsigned short* ldst) {
#if USE_ASYNC_LDS
  __builtin_amdgcn_global_load_async_to_lds_b128(
      (gas_v4i*)gsrc, (las_v4i*)ldst, 0, 0);
#else
  *(uint4*)ldst = *(const uint4*)gsrc;
#endif
}

__device__ __forceinline__ void stage_wait() {
#if USE_ASYNC_LDS
  __builtin_amdgcn_s_wait_asynccnt(0);
#endif
}

__device__ __forceinline__ unsigned short f32_to_bf16_rne(float f) {
  union { float f; unsigned u; } v; v.f = f;
  unsigned u = v.u;
  u += 0x7fffu + ((u >> 16) & 1u);
  return (unsigned short)(u >> 16);
}

__device__ __forceinline__ v16bf cat8(v8bf lo, v8bf hi) {
  return __builtin_shufflevector(lo, hi, 0,1,2,3,4,5,6,7,8,9,10,11,12,13,14,15);
}

// ---------------------------------------------------------------------------
// Kernel 1: convert hidden states to bf16, pack Wq|Wk|Wv -> [EE][NQK] bf16
// ---------------------------------------------------------------------------
__global__ void cvt_pack_kernel(const float* __restrict__ hs,
                                const float* __restrict__ Wq,
                                const float* __restrict__ Wk,
                                const float* __restrict__ Wv,
                                unsigned short* __restrict__ hsBF,
                                unsigned short* __restrict__ Wcat) {
  const int tid  = blockIdx.x * blockDim.x + threadIdx.x;
  const int nthr = gridDim.x * blockDim.x;
  const int n1 = BB * SS * EE;
  for (int i = tid; i < n1; i += nthr) hsBF[i] = f32_to_bf16_rne(hs[i]);
  const int n2 = EE * NQK;
  for (int i = tid; i < n2; i += nthr) {
    int k = i / NQK, j = i - k * NQK;
    float v;
    if (j < EE)          v = Wq[k * EE + j];
    else if (j < 2 * EE) v = Wk[k * EE + (j - EE)];
    else                 v = Wv[k * EE + (j - 2 * EE)];
    Wcat[i] = f32_to_bf16_rne(v);
  }
}

// ---------------------------------------------------------------------------
// Kernel 2: fused QKV GEMM  [8192 x 768] x [768 x 2304] in bf16 WMMA
// block tile 64(M) x 128(N), 8 waves of 32x32, K stepped by 32, double-buffered
// A tile staged via async global->LDS; B tile staged with in-flight transpose.
// ---------------------------------------------------------------------------
__launch_bounds__(256)
__global__ void qkv_gemm_kernel(const unsigned short* __restrict__ A,   // [8192][768] bf16
                                const unsigned short* __restrict__ Bw,  // [768][2304] bf16
                                const float* __restrict__ bq,
                                const float* __restrict__ bk,
                                const float* __restrict__ bv,
                                unsigned short* __restrict__ qBF,
                                unsigned short* __restrict__ kBF,
                                unsigned short* __restrict__ vBF) {
  __shared__ __align__(16) unsigned short As[2][64][40];    // [buf][m][k]
  __shared__ __align__(16) unsigned short Bs[2][128][40];   // [buf][n][k] (transposed)

  const int n0 = blockIdx.x * 128;
  const int m0 = blockIdx.y * 64;
  const int t  = threadIdx.x;
  const int lane = t & 31;
  const int wid  = t >> 5;
  const int waveM = wid >> 2;   // 0..1
  const int waveN = wid & 3;    // 0..3
  const int lhi = lane >> 4, llo = lane & 15;

  const v8f zero8 = {0.f,0.f,0.f,0.f,0.f,0.f,0.f,0.f};
  v8f acc[2][2];
  #pragma unroll
  for (int i = 0; i < 2; ++i)
    #pragma unroll
    for (int j = 0; j < 2; ++j) acc[i][j] = zero8;

  auto stageA = [&](int buf, int k0) {
    int row = t >> 2;            // 0..63
    int kc  = (t & 3) * 8;       // 0,8,16,24
    stage16B(A + (size_t)(m0 + row) * EE + k0 + kc, &As[buf][row][kc]);
  };
  auto stageB = [&](int buf, int k0) {
    int krow = t >> 4;           // 0..15
    int ncol = (t & 15) * 8;     // 0..120
    #pragma unroll
    for (int rep = 0; rep < 2; ++rep) {
      int kk = krow + rep * 16;
      uint4 d = *(const uint4*)(Bw + (size_t)(k0 + kk) * NQK + n0 + ncol);
      __align__(16) unsigned short vals[8];
      *(uint4*)&vals[0] = d;
      #pragma unroll
      for (int i = 0; i < 8; ++i) Bs[buf][ncol + i][kk] = vals[i];  // transpose
    }
  };

  stageA(0, 0);
  stageB(0, 0);
  stage_wait();
  __syncthreads();

  for (int kt = 0; kt < 24; ++kt) {
    int cur = kt & 1, nxt = cur ^ 1;
    if (kt + 1 < 24) { stageA(nxt, (kt + 1) * 32); stageB(nxt, (kt + 1) * 32); }

    v16bf aF[2], bF[2];
    #pragma unroll
    for (int mf = 0; mf < 2; ++mf) {
      const unsigned short* p = &As[cur][waveM * 32 + mf * 16 + llo][lhi * 8];
      aF[mf] = cat8(*(const v8bf*)p, *(const v8bf*)(p + 16));
    }
    #pragma unroll
    for (int nf = 0; nf < 2; ++nf) {
      const unsigned short* p = &Bs[cur][waveN * 32 + nf * 16 + llo][lhi * 8];
      bF[nf] = cat8(*(const v8bf*)p, *(const v8bf*)(p + 16));
    }
    #pragma unroll
    for (int mf = 0; mf < 2; ++mf)
      #pragma unroll
      for (int nf = 0; nf < 2; ++nf)
        acc[mf][nf] = __builtin_amdgcn_wmma_f32_16x16x32_bf16(
            false, aF[mf], false, bF[nf], (short)0, acc[mf][nf], false, false);
    stage_wait();
    __syncthreads();
  }

  // Epilogue: bias add, Q scale (1/sqrt(64)=0.125), convert to bf16, scatter
  #pragma unroll
  for (int mf = 0; mf < 2; ++mf) {
    int mbase = m0 + waveM * 32 + mf * 16 + lhi * 8;
    #pragma unroll
    for (int nf = 0; nf < 2; ++nf) {
      int n = n0 + waveN * 32 + nf * 16 + llo;
      int which = n / EE;
      int nn = n - which * EE;
      #pragma unroll
      for (int r = 0; r < 8; ++r) {
        float val = acc[mf][nf][r];
        unsigned short* dst;
        if (which == 0)      { val = (val + bq[nn]) * 0.125f; dst = qBF; }
        else if (which == 1) { val =  val + bk[nn];           dst = kBF; }
        else                 { val =  val + bv[nn];           dst = vBF; }
        dst[(size_t)(mbase + r) * EE + nn] = f32_to_bf16_rne(val);
      }
    }
  }
}

// ---------------------------------------------------------------------------
// Kernel 3: banded flash attention. One WG per (chunk c, head h, batch b).
// 8 waves x 32 query rows; 12 key blocks of 64 keys covering the 3W window.
// Out-of-window keys are CLAMPED (not zero-filled): the band mask forces their
// probabilities to exactly 0, so arbitrary finite K/V data is harmless.
// ---------------------------------------------------------------------------
__launch_bounds__(256)
__global__ void attn_kernel(const unsigned short* __restrict__ qBF,
                            const unsigned short* __restrict__ kBF,
                            const unsigned short* __restrict__ vBF,
                            const float* __restrict__ amask,   // [B, S]
                            float* __restrict__ out) {
  __shared__ __align__(16) unsigned short Kblk[64][72];       // [key][d]
  __shared__ __align__(16) unsigned short Vt[64][72];         // [d][key]
  __shared__ __align__(16) unsigned short Pbuf[8][32][72];    // per-wave P tiles

  const int c = blockIdx.x;
  const int h = blockIdx.y;
  const int b = blockIdx.z;
  const int t = threadIdx.x;
  const int lane = t & 31, wid = t >> 5;
  const int lhi = lane >> 4, llo = lane & 15;

  // Load the wave's Q fragments once (A-layout: contiguous 16B per half-row)
  v16bf aQ[2][2];
  #pragma unroll
  for (int mf = 0; mf < 2; ++mf) {
    int s = c * WW + wid * 32 + mf * 16 + llo;
    const unsigned short* qrow = qBF + (size_t)(b * SS + s) * EE + h * DD;
    #pragma unroll
    for (int kt = 0; kt < 2; ++kt) {
      const unsigned short* p = qrow + kt * 32 + lhi * 8;
      aQ[mf][kt] = cat8(*(const v8bf*)p, *(const v8bf*)(p + 16));
    }
  }

  const v8f zero8 = {0.f,0.f,0.f,0.f,0.f,0.f,0.f,0.f};
  v8f Oacc[2][4];
  float rowMax[2][8], rowSum[2][8];
  #pragma unroll
  for (int mf = 0; mf < 2; ++mf) {
    #pragma unroll
    for (int nd = 0; nd < 4; ++nd) Oacc[mf][nd] = zero8;
    #pragma unroll
    for (int r = 0; r < 8; ++r) { rowMax[mf][r] = -1e30f; rowSum[mf][r] = 0.f; }
  }

  for (int kb = 0; kb < 12; ++kb) {
    __syncthreads();
    // ---- stage K block (row major) via async global->LDS ----
    #pragma unroll
    for (int rep = 0; rep < 2; ++rep) {
      int u = t + rep * 256;
      int key = u >> 3, chunk = u & 7;
      int skey = c * WW + kb * 64 + key - WW;
      int scl = skey < 0 ? 0 : (skey >= SS ? SS - 1 : skey);   // clamp: masked anyway
      stage16B(kBF + (size_t)(b * SS + scl) * EE + h * DD + chunk * 8,
               &Kblk[key][chunk * 8]);
    }
    // ---- stage V block transposed (manual scatter) ----
    {
      int key = t >> 2;
      int d0  = (t & 3) * 16;
      int skey = c * WW + kb * 64 + key - WW;
      int scl = skey < 0 ? 0 : (skey >= SS ? SS - 1 : skey);
      const unsigned short* vp = vBF + (size_t)(b * SS + scl) * EE + h * DD + d0;
      uint4 d0v = *(const uint4*)vp;
      uint4 d1v = *(const uint4*)(vp + 8);
      if (kb + 1 < 12 && skey + 64 >= 0 && skey + 64 < SS)
        __builtin_prefetch(vp + (size_t)64 * EE, 0, 0);        // next V block
      __align__(16) unsigned short vals[16];
      *(uint4*)&vals[0] = d0v;
      *(uint4*)&vals[8] = d1v;
      #pragma unroll
      for (int i = 0; i < 16; ++i) Vt[d0 + i][key] = vals[i];
    }
    stage_wait();
    __syncthreads();

    // ---- scores S = Q * K^T  (32 rows x 64 keys per wave) ----
    v8f sAcc[2][4];
    #pragma unroll
    for (int mf = 0; mf < 2; ++mf)
      #pragma unroll
      for (int nf = 0; nf < 4; ++nf) sAcc[mf][nf] = zero8;

    #pragma unroll
    for (int kt = 0; kt < 2; ++kt) {
      v16bf bK[4];
      #pragma unroll
      for (int nf = 0; nf < 4; ++nf) {
        const unsigned short* p = &Kblk[nf * 16 + llo][kt * 32 + lhi * 8];
        bK[nf] = cat8(*(const v8bf*)p, *(const v8bf*)(p + 16));
      }
      #pragma unroll
      for (int mf = 0; mf < 2; ++mf)
        #pragma unroll
        for (int nf = 0; nf < 4; ++nf)
          sAcc[mf][nf] = __builtin_amdgcn_wmma_f32_16x16x32_bf16(
              false, aQ[mf][kt], false, bK[nf], (short)0, sAcc[mf][nf], false, false);
    }

    // key bias / bounds per nf (per-lane key column)
    float kbias[4];
    int olane[4];
    #pragma unroll
    for (int nf = 0; nf < 4; ++nf) {
      int o = kb * 64 + nf * 16 + llo;
      olane[nf] = o;
      int skey = c * WW + o - WW;
      float bia = -1e30f;                       // out-of-sequence -> masked
      if (skey >= 0 && skey < SS) {
        float mv = amask[b * SS + skey];
        bia = (mv != 0.f) ? -10000.f : 0.f;     // diagonal key bias
      }
      kbias[nf] = bia;
    }

    // ---- online softmax (per half-wave row groups) ----
    #pragma unroll
    for (int mf = 0; mf < 2; ++mf) {
      float pm[8];
      #pragma unroll
      for (int r = 0; r < 8; ++r) pm[r] = -1e30f;
      #pragma unroll
      for (int nf = 0; nf < 4; ++nf) {
        #pragma unroll
        for (int r = 0; r < 8; ++r) {
          int rq = wid * 32 + mf * 16 + lhi * 8 + r;  // query row in chunk
          int diff = olane[nf] - WW - rq;
          float s = sAcc[mf][nf][r] + kbias[nf];
          if (diff < -WW || diff > WW) s = -1e30f;    // band mask |j-i|<=W
          if (s < -1e29f) s = -1e30f;
          sAcc[mf][nf][r] = s;
          pm[r] = fmaxf(pm[r], s);
        }
      }
      #pragma unroll
      for (int r = 0; r < 8; ++r) {
        float v = pm[r];
        v = fmaxf(v, __shfl_xor(v, 1, 32));
        v = fmaxf(v, __shfl_xor(v, 2, 32));
        v = fmaxf(v, __shfl_xor(v, 4, 32));
        v = fmaxf(v, __shfl_xor(v, 8, 32));
        pm[r] = v;
      }
      float corr[8];
      #pragma unroll
      for (int r = 0; r < 8; ++r) {
        float nm = fmaxf(rowMax[mf][r], pm[r]);
        corr[r] = __expf(rowMax[mf][r] - nm);
        rowMax[mf][r] = nm;
      }
      #pragma unroll
      for (int nd = 0; nd < 4; ++nd)
        #pragma unroll
        for (int r = 0; r < 8; ++r)
          Oacc[mf][nd][r] = Oacc[mf][nd][r] * corr[r];
      float ps[8];
      #pragma unroll
      for (int r = 0; r < 8; ++r) ps[r] = 0.f;
      #pragma unroll
      for (int nf = 0; nf < 4; ++nf) {
        #pragma unroll
        for (int r = 0; r < 8; ++r) {
          float s = sAcc[mf][nf][r];
          float p = (s < -1e29f) ? 0.f : __expf(s - rowMax[mf][r]);
          ps[r] += p;
          Pbuf[wid][mf * 16 + lhi * 8 + r][nf * 16 + llo] = f32_to_bf16_rne(p);
        }
      }
      #pragma unroll
      for (int r = 0; r < 8; ++r) {
        float v = ps[r];
        v += __shfl_xor(v, 1, 32);
        v += __shfl_xor(v, 2, 32);
        v += __shfl_xor(v, 4, 32);
        v += __shfl_xor(v, 8, 32);
        rowSum[mf][r] = rowSum[mf][r] * corr[r] + v;
      }
    }

    // ---- O += P * V ----
    #pragma unroll
    for (int kt = 0; kt < 2; ++kt) {
      v16bf aP[2], bV[4];
      #pragma unroll
      for (int mf = 0; mf < 2; ++mf) {
        const unsigned short* p = &Pbuf[wid][mf * 16 + llo][kt * 32 + lhi * 8];
        aP[mf] = cat8(*(const v8bf*)p, *(const v8bf*)(p + 16));
      }
      #pragma unroll
      for (int nd = 0; nd < 4; ++nd) {
        const unsigned short* p = &Vt[nd * 16 + llo][kt * 32 + lhi * 8];
        bV[nd] = cat8(*(const v8bf*)p, *(const v8bf*)(p + 16));
      }
      #pragma unroll
      for (int mf = 0; mf < 2; ++mf)
        #pragma unroll
        for (int nd = 0; nd < 4; ++nd)
          Oacc[mf][nd] = __builtin_amdgcn_wmma_f32_16x16x32_bf16(
              false, aP[mf], false, bV[nd], (short)0, Oacc[mf][nd], false, false);
    }
  }

  // ---- epilogue: normalize, zero masked queries, store f32 ----
  #pragma unroll
  for (int mf = 0; mf < 2; ++mf) {
    #pragma unroll
    for (int r = 0; r < 8; ++r) {
      int mloc = wid * 32 + mf * 16 + lhi * 8 + r;
      int s = c * WW + mloc;
      float qm = amask[b * SS + s];
      float inv = (qm < 0.f) ? 0.f : (1.0f / rowSum[mf][r]);
      float* orow = out + (size_t)(b * SS + s) * EE + h * DD;
      #pragma unroll
      for (int nd = 0; nd < 4; ++nd)
        orow[nd * 16 + llo] = Oacc[mf][nd][r] * inv;
    }
  }
}

// ---------------------------------------------------------------------------
extern "C" void kernel_launch(void* const* d_in, const int* in_sizes, int n_in,
                              void* d_out, int out_size, void* d_ws, size_t ws_size,
                              hipStream_t stream) {
  const float* hs    = (const float*)d_in[0];
  const float* amask = (const float*)d_in[1];
  const float* Wq    = (const float*)d_in[2];
  const float* bq    = (const float*)d_in[3];
  const float* Wk    = (const float*)d_in[4];
  const float* bk    = (const float*)d_in[5];
  const float* Wv    = (const float*)d_in[6];
  const float* bv    = (const float*)d_in[7];
  float* out = (float*)d_out;

  char* ws = (char*)d_ws;
  size_t off = 0;
  auto take = [&](size_t bytes) -> char* {
    char* p = ws + off;
    off = (off + bytes + 255) & ~(size_t)255;
    return p;
  };
  unsigned short* hsBF = (unsigned short*)take((size_t)BB * SS * EE * 2);
  unsigned short* Wcat = (unsigned short*)take((size_t)EE * NQK * 2);
  unsigned short* qBF  = (unsigned short*)take((size_t)BB * SS * EE * 2);
  unsigned short* kBF  = (unsigned short*)take((size_t)BB * SS * EE * 2);
  unsigned short* vBF  = (unsigned short*)take((size_t)BB * SS * EE * 2);

  cvt_pack_kernel<<<1024, 256, 0, stream>>>(hs, Wq, Wk, Wv, hsBF, Wcat);
  qkv_gemm_kernel<<<dim3(NQK / 128, (BB * SS) / 64), 256, 0, stream>>>(
      hsBF, Wcat, bq, bk, bv, qBF, kBF, vBF);
  attn_kernel<<<dim3(NCHUNK, HH, BB), 256, 0, stream>>>(qBF, kBF, vBF, amask, out);
}
